// GatedDeltaNetLayer_59322088293086
// MI455X (gfx1250) — matrix operand order repaired and checked by reading, not compiled, not run on hardware
//
#include <hip/hip_runtime.h>
#include <hip/hip_bf16.h>

// ---------------------------------------------------------------------------
// GatedDeltaNet prefill for MI455X (gfx1250, wave32).
// GEMMs: v_wmma_f32_16x16x32_bf16 (f32 accum). ~150MB unique HBM traffic
// @23.3TB/s dominates; bf16 WMMA keeps compute far under memory time.
// Scan: sequential over S -> latency-bound; k/q/v double-buffered into LDS
// with global_load_async_to_lds_b64 + s_wait_asynccnt to hide L2 latency
// on the serial chain.
// ---------------------------------------------------------------------------

#define SEQ   2048
#define HID   2048
#define NHEAD 16
#define DHEAD 128
#define CONVC 6144
#define DKT   2048
#define DVT   2048
#define EPSV  1e-6f

typedef __attribute__((ext_vector_type(16))) __bf16 v16bf;
typedef __attribute__((ext_vector_type(8)))  __bf16 v8bf;
typedef __attribute__((ext_vector_type(4)))  __bf16 v4bf;
typedef __attribute__((ext_vector_type(8)))  float  v8f;
typedef __attribute__((ext_vector_type(4)))  float  f4;

__device__ __forceinline__ __bf16 cvt_bf16(float f) {
  unsigned u = __builtin_bit_cast(unsigned, f);
  unsigned r = u + 0x7FFFu + ((u >> 16) & 1u);      // round-to-nearest-even
  unsigned short h = (unsigned short)(r >> 16);
  return __builtin_bit_cast(__bf16, h);
}

// ---------------------------------------------------------------------------
// Generic C(MxN) = W(MxK) * X(KxN), all f32 row-major in memory, bf16 WMMA.
// Block: 256 thr (8 waves). Tile 128x128, K-step 32 (= one WMMA K).
// Wave (wm,wn) in 2x4 grid computes a 64x32 region = 4x2 WMMA tiles.
// ---------------------------------------------------------------------------
#define BM 128
#define BN 128
#define BK 32
#define LDA 40   // padded halves per A-tile row  (80B rows, 16B aligned frags)
#define LDB 40   // padded halves per Bt-tile row (col-major B: [n][k])

__global__ __launch_bounds__(256)
void gemm_bf16_wmma(const float* __restrict__ W, const float* __restrict__ X,
                    float* __restrict__ C, int M, int N, int K) {
  __shared__ __align__(16) __bf16 As[BM * LDA];
  __shared__ __align__(16) __bf16 Bs[BN * LDB];

  const int tid  = threadIdx.x;
  const int lane = tid & 31;
  const int wid  = tid >> 5;
  const int wm   = wid >> 2;     // 0..1
  const int wn   = wid & 3;      // 0..3
  const int m0   = blockIdx.x * BM;
  const int n0   = blockIdx.y * BN;

  v8f acc[4][2];
#pragma unroll
  for (int i = 0; i < 4; ++i)
#pragma unroll
    for (int j = 0; j < 2; ++j) {
      v8f z = {0.f,0.f,0.f,0.f,0.f,0.f,0.f,0.f};
      acc[i][j] = z;
    }

  for (int k0 = 0; k0 < K; k0 += BK) {
    __syncthreads();
    // ---- A tile: 128x32 f32 -> bf16 LDS (row-major). 4 x float4 per thread.
#pragma unroll
    for (int i = 0; i < 4; ++i) {
      int idx = tid + i * 256;
      int r  = idx >> 3;
      int c4 = (idx & 7) << 2;
      f4 v = {0.f, 0.f, 0.f, 0.f};
      if (m0 + r < M) v = *(const f4*)(W + (size_t)(m0 + r) * K + k0 + c4);
      v4bf pb;
      pb[0] = cvt_bf16(v[0]); pb[1] = cvt_bf16(v[1]);
      pb[2] = cvt_bf16(v[2]); pb[3] = cvt_bf16(v[3]);
      *(v4bf*)(&As[r * LDA + c4]) = pb;
    }
    // ---- B tile: 32x128 f32 -> bf16 LDS *transposed* ([n][k], K-contiguous)
#pragma unroll
    for (int i = 0; i < 4; ++i) {
      int idx = tid + i * 256;
      int kk = idx >> 5;
      int n4 = (idx & 31) << 2;
      f4 v = *(const f4*)(X + (size_t)(k0 + kk) * N + n0 + n4);
      Bs[(n4 + 0) * LDB + kk] = cvt_bf16(v[0]);
      Bs[(n4 + 1) * LDB + kk] = cvt_bf16(v[1]);
      Bs[(n4 + 2) * LDB + kk] = cvt_bf16(v[2]);
      Bs[(n4 + 3) * LDB + kk] = cvt_bf16(v[3]);
    }
    // prefetch next K-tile into cache hierarchy (global_prefetch_b8)
    if (k0 + BK < K) {
      int r = tid >> 1;
      if (m0 + r < M)
        __builtin_prefetch(W + (size_t)(m0 + r) * K + k0 + BK, 0, 1);
      __builtin_prefetch(X + (size_t)(k0 + BK + (tid >> 4)) * N + n0 + ((tid & 15) << 3), 0, 1);
    }
    __syncthreads();

    // ---- A fragments (ISA 16-bit A 16x32 layout):
    // lanes 0-15: halves 0..7 = K 0..7, halves 8..15 = K 16..23
    // lanes16-31: halves 0..7 = K 8..15, halves 8..15 = K 24..31
    const int arow = wm * 64 + (lane & 15);
    const int kA   = (lane < 16) ? 0 : 8;
    v16bf afrag[4];
#pragma unroll
    for (int i = 0; i < 4; ++i) {
      const __bf16* p = &As[(arow + i * 16) * LDA];
      v8bf lo = *(const v8bf*)(p + kA);
      v8bf hi = *(const v8bf*)(p + kA + 16);
      afrag[i] = __builtin_shufflevector(lo, hi, 0,1,2,3,4,5,6,7,8,9,10,11,12,13,14,15);
    }
    // ---- B fragments: lane = N col; lanes 0-15 K 0..7,16..23, lanes 16-31 K 8..15,24..31
    const int kB = (lane < 16) ? 0 : 8;
    v16bf bfrag[2];
#pragma unroll
    for (int j = 0; j < 2; ++j) {
      const __bf16* p = &Bs[(wn * 32 + j * 16 + (lane & 15)) * LDB + kB];
      v8bf lo = *(const v8bf*)(p);
      v8bf hi = *(const v8bf*)(p + 16);
      bfrag[j] = __builtin_shufflevector(lo, hi, 0,1,2,3,4,5,6,7,8,9,10,11,12,13,14,15);
    }
#pragma unroll
    for (int i = 0; i < 4; ++i)
#pragma unroll
      for (int j = 0; j < 2; ++j)
        acc[i][j] = __builtin_amdgcn_wmma_f32_16x16x32_bf16(
            false, afrag[i], false, bfrag[j], (short)0, acc[i][j], false, false);
  }

  // ---- store (C/D layout: VGPR r -> M = r + 8*(lane>=16), lane&15 -> N)
  const int colBase = n0 + wn * 32;
#pragma unroll
  for (int i = 0; i < 4; ++i) {
    int rowBase = m0 + wm * 64 + i * 16 + ((lane >> 4) << 3);
#pragma unroll
    for (int j = 0; j < 2; ++j) {
      int col = colBase + j * 16 + (lane & 15);
#pragma unroll
      for (int r = 0; r < 8; ++r) {
        int row = rowBase + r;
        if (row < M) C[(size_t)row * N + col] = acc[i][j][r];
      }
    }
  }
}

// ---------------------------------------------------------------------------
// Depthwise causal conv (KS=4) + SiLU; also emits new_conv_cache.
// grid (SEQ/256, CONVC), conv_in = [cache(3) | qkv(SEQ)] per channel.
// ---------------------------------------------------------------------------
__global__ __launch_bounds__(256)
void conv_silu_kernel(const float* __restrict__ qkv, const float* __restrict__ cache_in,
                      const float* __restrict__ Wc, float* __restrict__ out,
                      float* __restrict__ cache_out) {
  const int s = blockIdx.x * 256 + threadIdx.x;
  const int c = blockIdx.y;
  float w[4];
#pragma unroll
  for (int j = 0; j < 4; ++j) {
    int i = s + j;
    w[j] = (i < 3) ? cache_in[c * 3 + i] : qkv[(size_t)c * SEQ + (i - 3)];
  }
  float a = 0.f;
#pragma unroll
  for (int j = 0; j < 4; ++j) a += Wc[c * 4 + j] * w[j];
  out[(size_t)c * SEQ + s] = a / (1.f + __expf(-a));
  if (s >= SEQ - 3) cache_out[c * 3 + (s - (SEQ - 3))] = w[3];  // qkv[:, S-3:]
}

// ---------------------------------------------------------------------------
// RMS-norm q/k (+scales), reshape to (s,h,d); compute g, beta.
// grid (SEQ, NHEAD), 128 threads (d).
// ---------------------------------------------------------------------------
__global__ __launch_bounds__(128)
void qkv_norm_kernel(const float* __restrict__ conv_out, const float* __restrict__ a_pre,
                     const float* __restrict__ b_pre, const float* __restrict__ A_log,
                     const float* __restrict__ dt_bias,
                     float* __restrict__ qn, float* __restrict__ kn, float* __restrict__ vn,
                     float* __restrict__ g, float* __restrict__ beta) {
  __shared__ float red[8];
  const int s = blockIdx.x, h = blockIdx.y, d = threadIdx.x;
  float qv = conv_out[(size_t)(h * DHEAD + d) * SEQ + s];
  float kv = conv_out[(size_t)(DKT + h * DHEAD + d) * SEQ + s];
  float vv = conv_out[(size_t)(2 * DKT + h * DHEAD + d) * SEQ + s];
  float pq = qv * qv, pk = kv * kv;
  for (int off = 16; off; off >>= 1) {
    pq += __shfl_xor(pq, off, 32);
    pk += __shfl_xor(pk, off, 32);
  }
  const int wv = threadIdx.x >> 5;
  if ((threadIdx.x & 31) == 0) { red[wv] = pq; red[4 + wv] = pk; }
  __syncthreads();
  float msq = (red[0] + red[1] + red[2] + red[3]) * (1.f / 128.f);
  float msk = (red[4] + red[5] + red[6] + red[7]) * (1.f / 128.f);
  const float inv = 0.08838834764831845f;  // DK^-0.5
  const size_t o = (size_t)s * DVT + h * DHEAD + d;
  qn[o] = (inv * inv) * qv * rsqrtf(msq + EPSV);
  kn[o] = inv * kv * rsqrtf(msk + EPSV);
  vn[o] = vv;
  if (d == 0) {
    float A  = __expf(A_log[0]);
    float xx = a_pre[(size_t)h * SEQ + s] + dt_bias[0];
    float sp = (xx > 20.f) ? xx : log1pf(__expf(xx));
    g[s * NHEAD + h]    = __expf(-A * sp);
    float bb = b_pre[(size_t)h * SEQ + s];
    beta[s * NHEAD + h] = 1.f / (1.f + __expf(-bb));
  }
}

// ---------------------------------------------------------------------------
// Async k|q|v stage: threads 0..191 each move 8B global->LDS (1536B total)
// via the CDNA5 async-DMA path (tracked by ASYNCcnt, not LOADcnt).
// LDS operand = low 32 bits of the flat shared-pointer (ISA 10.2: LDS
// aperture truncates flat addr to addr[31:0]).
// ---------------------------------------------------------------------------
__device__ __forceinline__ void issue_kqv_async(const float* kp, const float* qp,
                                                const float* vp, float* dst, int tid) {
  const int e = tid * 2;
  if (e < 384) {
    const float* src = (e < 128) ? (kp + e)
                     : (e < 256) ? (qp + (e - 128))
                                 : (vp + (e - 256));
    unsigned lds = (unsigned)(unsigned long long)(dst + e);
    asm volatile("global_load_async_to_lds_b64 %0, %1, off"
                 :: "v"(lds), "v"(src) : "memory");
  }
}

// ---------------------------------------------------------------------------
// Sequential delta-rule scan. 1 block/head, 256 threads (8 waves).
// Wave w owns v-rows [16w,16w+16); lane owns k = lane+32j (j<4).
// State in registers: 64 f32/lane. k/q/v for step s+1 are async-DMA'd into
// the alternate LDS buffer while step s computes; s_wait_asynccnt 1 retires
// the older batch (async loads complete in order per wave).
// ---------------------------------------------------------------------------
__global__ __launch_bounds__(256)
void delta_scan_kernel(const float* __restrict__ qn, const float* __restrict__ kn,
                       const float* __restrict__ vn, const float* __restrict__ g,
                       const float* __restrict__ beta, const float* __restrict__ state_in,
                       float* __restrict__ ys, float* __restrict__ state_out) {
  __shared__ __align__(16) float kqv[2][384];   // [buf][ k(128) | q(128) | v(128) ]
  const int h    = blockIdx.x;
  const int tid  = threadIdx.x;
  const int lane = tid & 31;
  const int wv   = tid >> 5;
  float st[16][4];
#pragma unroll
  for (int r = 0; r < 16; ++r)
#pragma unroll
    for (int j = 0; j < 4; ++j)
      st[r][j] = state_in[((size_t)h * 128 + (wv * 16 + r)) * 128 + lane + 32 * j];

  // prologue: stage step 0
  issue_kqv_async(kn + (size_t)0 * DVT + h * DHEAD,
                  qn + (size_t)0 * DVT + h * DHEAD,
                  vn + (size_t)0 * DVT + h * DHEAD, kqv[0], tid);

  for (int s = 0; s < SEQ; ++s) {
    if (s + 1 < SEQ) {
      issue_kqv_async(kn + (size_t)(s + 1) * DVT + h * DHEAD,
                      qn + (size_t)(s + 1) * DVT + h * DHEAD,
                      vn + (size_t)(s + 1) * DVT + h * DHEAD,
                      kqv[(s + 1) & 1], tid);
      asm volatile("s_wait_asynccnt 0x1" ::: "memory");  // step-s batch resident
    } else {
      asm volatile("s_wait_asynccnt 0x0" ::: "memory");  // final batch resident
    }
    __syncthreads();

    const float* buf = kqv[s & 1];
    float kk[4], qq[4];
#pragma unroll
    for (int j = 0; j < 4; ++j) {
      kk[j] = buf[lane + 32 * j];
      qq[j] = buf[128 + lane + 32 * j];
    }
    const float gs = g[s * NHEAD + h];
    const float bs = beta[s * NHEAD + h];
#pragma unroll
    for (int r = 0; r < 16; ++r) {
      float p = st[r][0] * kk[0] + st[r][1] * kk[1] + st[r][2] * kk[2] + st[r][3] * kk[3];
      for (int off = 16; off; off >>= 1) p += __shfl_xor(p, off, 32);
      float vt = buf[256 + wv * 16 + r];
      float delta = (vt - gs * p) * bs;
#pragma unroll
      for (int j = 0; j < 4; ++j) st[r][j] = st[r][j] * gs + delta * kk[j];
      float y = st[r][0] * qq[0] + st[r][1] * qq[1] + st[r][2] * qq[2] + st[r][3] * qq[3];
      for (int off = 16; off; off >>= 1) y += __shfl_xor(y, off, 32);
      if (lane == r) ys[(size_t)s * DVT + h * DHEAD + wv * 16 + r] = y;
    }
    __syncthreads();  // readers done before next iter's async writes reuse buf
  }
#pragma unroll
  for (int r = 0; r < 16; ++r)
#pragma unroll
    for (int j = 0; j < 4; ++j)
      state_out[((size_t)h * 128 + (wv * 16 + r)) * 128 + lane + 32 * j] = st[r][j];
}

// ---------------------------------------------------------------------------
// y_gated = norm_w * rms_norm(ys) * silu(z); store transposed (DVT, SEQ) so
// it is directly the B operand of the final GEMM. grid (SEQ, NHEAD), 128 thr.
// ---------------------------------------------------------------------------
__global__ __launch_bounds__(128)
void gate_kernel(const float* __restrict__ ys, const float* __restrict__ z_pre,
                 const float* __restrict__ norm_w, float* __restrict__ ygT) {
  __shared__ float red[4];
  const int s = blockIdx.x, h = blockIdx.y, d = threadIdx.x;
  float y = ys[(size_t)s * DVT + h * DHEAD + d];
  float p = y * y;
  for (int off = 16; off; off >>= 1) p += __shfl_xor(p, off, 32);
  if ((threadIdx.x & 31) == 0) red[threadIdx.x >> 5] = p;
  __syncthreads();
  float ms = (red[0] + red[1] + red[2] + red[3]) * (1.f / 128.f);
  float zz = z_pre[(size_t)(h * DHEAD + d) * SEQ + s];
  float silu = zz / (1.f + __expf(-zz));
  ygT[(size_t)(h * DHEAD + d) * SEQ + s] = norm_w[d] * y * rsqrtf(ms + EPSV) * silu;
}

// ---------------------------------------------------------------------------
extern "C" void kernel_launch(void* const* d_in, const int* in_sizes, int n_in,
                              void* d_out, int out_size, void* d_ws, size_t ws_size,
                              hipStream_t stream) {
  const float* x      = (const float*)d_in[0];   // (H, S)
  const float* dstate = (const float*)d_in[1];   // (16,128,128)
  const float* ccache = (const float*)d_in[2];   // (6144, 3)
  /* d_in[3] = position (unused by the math) */
  const float* W_qkv  = (const float*)d_in[4];
  const float* W_a    = (const float*)d_in[5];
  const float* W_b    = (const float*)d_in[6];
  const float* W_z    = (const float*)d_in[7];
  const float* W_out  = (const float*)d_in[8];
  const float* W_conv = (const float*)d_in[9];
  const float* A_log  = (const float*)d_in[10];
  const float* dt_b   = (const float*)d_in[11];
  const float* norm_w = (const float*)d_in[12];

  float* out_y     = (float*)d_out;                  // (2048, 2048)
  float* out_state = out_y + (size_t)HID * SEQ;      // 16*128*128
  float* out_cache = out_state + 16 * 128 * 128;     // 6144*3

  // Workspace (floats), with lifetime-based aliasing: total ~118 MB.
  float* ws      = (float*)d_ws;
  float* qkv_pre = ws;                    // 12,582,912  -> later qn/kn/vn
  float* convb   = ws + 12582912;         // 12,582,912  -> later ys/ygT
  float* z_pre   = ws + 25165824;         //  4,194,304
  float* a_pre   = ws + 29360128;         //     32,768
  float* b_pre   = a_pre + 32768;         //     32,768
  float* gbuf    = b_pre + 32768;         //     32,768
  float* betab   = gbuf + 32768;          //     32,768
  float* qn  = qkv_pre;                   // alias after conv consumed qkv_pre
  float* kn  = qkv_pre + 4194304;
  float* vn  = qkv_pre + 8388608;
  float* ysb = convb;                     // alias after norm consumed conv_out
  float* ygT = convb + 4194304;

  dim3 blk256(256), blk128(128);

  // 1) Projections (bf16 WMMA GEMMs)
  gemm_bf16_wmma<<<dim3(48, 16), blk256, 0, stream>>>(W_qkv, x, qkv_pre, 6144, 2048, 2048);
  gemm_bf16_wmma<<<dim3(1, 16),  blk256, 0, stream>>>(W_a,   x, a_pre,   16,   2048, 2048);
  gemm_bf16_wmma<<<dim3(1, 16),  blk256, 0, stream>>>(W_b,   x, b_pre,   16,   2048, 2048);
  gemm_bf16_wmma<<<dim3(16, 16), blk256, 0, stream>>>(W_z,   x, z_pre,   2048, 2048, 2048);
  // 2) conv + SiLU + new cache
  conv_silu_kernel<<<dim3(8, CONVC), blk256, 0, stream>>>(qkv_pre, ccache, W_conv, convb, out_cache);
  // 3) q/k RMS-norm + gates
  qkv_norm_kernel<<<dim3(SEQ, NHEAD), blk128, 0, stream>>>(convb, a_pre, b_pre, A_log, dt_b,
                                                           qn, kn, vn, gbuf, betab);
  // 4) recurrent scan (sequential over S, async-staged k/q/v)
  delta_scan_kernel<<<dim3(NHEAD), blk256, 0, stream>>>(qn, kn, vn, gbuf, betab, dstate,
                                                        ysb, out_state);
  // 5) gated output norm (stores B operand transposed)
  gate_kernel<<<dim3(SEQ, NHEAD), blk128, 0, stream>>>(ysb, z_pre, norm_w, ygT);
  // 6) output projection
  gemm_bf16_wmma<<<dim3(16, 16), blk256, 0, stream>>>(W_out, ygT, out_y, 2048, 2048, 2048);
}
